// TileInitialization_64802466562435
// MI455X (gfx1250) — compile-verified, matrix operand order checked by reading, not run on hardware
//
#include <hip/hip_runtime.h>
#include <hip/hip_bf16.h>

typedef __attribute__((ext_vector_type(16))) _Float16 v16h;
typedef __attribute__((ext_vector_type(8)))  float    v8f;

__device__ __forceinline__ float lrelu(float x) { return x >= 0.f ? x : 0.2f * x; }

// K-offset within a 32-chunk for fragment half-word slot hv (0..15), per lane half.
// CDNA5 16-bit A-matrix 16x32 layout: VGPR r (=hv>>1), low/high half (=hv&1):
//   r 0..3 -> K = (laneHi?8:0) + 2r + h ; r 4..7 -> K = 16 + (laneHi?8:0) + 2(r-4) + h
__device__ __forceinline__ int kmap(int laneHi, int hv) {
    int r = hv >> 1, h = hv & 1;
    return ((r >= 4) ? 16 : 0) + (laneHi ? 8 : 0) + 2 * (r & 3) + h;
}

#define NWAVES 8     // 256-thread block = 8 wave32
#define NT 2         // N-tiles (of 16 pixels) per wave

// Fused: 4x4 conv (stride 4 x strideX) + bias + lrelu + 1x1 conv (16x16) + bias + lrelu.
// Each wave computes a row-aligned 32-pixel x 16-channel output strip via 2 WMMAs per
// 32-wide K chunk (A-fragment shared). Weights pre-swizzled into LDS as f16 fragments.
// Output stored HWC: out[(oy*Wout+ox)*16 + c]. Reads with ix >= Win return 0 (right pad).
__global__ void __launch_bounds__(256)
conv_tile_wmma(const float* __restrict__ in, int Cin, int Hin, int Win, int strideX,
               const float* __restrict__ wa, const float* __restrict__ ba,
               const float* __restrict__ wb, const float* __restrict__ bb,
               float* __restrict__ out, int Hout, int Wout)
{
    // wsm: conv-A weights in fragment order, (chunk, lane) -> 16 halves (32B each)
    __shared__ __align__(32) _Float16 wsm[16 * 32 * 16];   // up to K=512 -> 16 chunks
    __shared__ __align__(32) _Float16 wbsm[32 * 16];       // 1x1 weights, K padded to 32
    __shared__ float act[NWAVES * NT * 256];               // per-wave 16x16 f32 act tiles

    const int tid    = threadIdx.x;
    const int lane   = tid & 31;
    const int wid    = tid >> 5;
    const int laneHi = lane >> 4;
    const int nIdx   = lane & 15;          // M for A / N for B,C,D
    const int K      = Cin * 16;
    const int nChunks = K >> 5;

    // ---- stage conv-A weights into LDS in fragment order (f32 -> f16) ----
    for (int e = tid; e < nChunks * 32; e += 256) {
        const int ch = e >> 5, ln = e & 31;
        const int lh = ln >> 4, m = ln & 15;
        #pragma unroll
        for (int hv = 0; hv < 16; ++hv) {
            const int k = ch * 32 + kmap(lh, hv);
            wsm[e * 16 + hv] = (_Float16)wa[m * K + k];    // wa flat: m*K + c*16+ky*4+kx
        }
    }
    // ---- stage 1x1 weights (K 16 -> padded 32) ----
    if (tid < 32) {
        const int lh = tid >> 4, m = tid & 15;
        #pragma unroll
        for (int hv = 0; hv < 16; ++hv) {
            const int k = kmap(lh, hv);
            wbsm[tid * 16 + hv] = (_Float16)((k < 16) ? wb[m * 16 + k] : 0.f);
        }
    }
    __syncthreads();

    // ---- which 32-pixel row strip does this wave own ----
    const int tilesPerRow = (Wout + NT * 16 - 1) / (NT * 16);
    const int tiles = Hout * tilesPerRow;
    const int tile  = blockIdx.x * NWAVES + wid;
    const bool tvalid = tile < tiles;
    const int ti  = tvalid ? tile : 0;
    const int oy  = ti / tilesPerRow;
    const int oxB = (ti % tilesPerRow) * (NT * 16);

    __builtin_prefetch(in + (oy * 4) * Win + oxB * strideX, 0, 1);

    // ---- precompute chunk-invariant gather offsets and pad masks ----
    // k-within-chunk = kmap(laneHi,hv): chi = high channel bit, (ky,kx) from low 4 bits
    int  off[NT][16];
    bool sel[NT][16];
    #pragma unroll
    for (int hv = 0; hv < 16; ++hv) {
        const int kk  = kmap(laneHi, hv);
        const int chi = kk >> 4;            // channel within the 2-channel chunk
        const int tt  = kk & 15;
        const int iy  = oy * 4 + (tt >> 2);
        const int kx  = tt & 3;
        const int rowBase = (chi * Hin + iy) * Win;
        #pragma unroll
        for (int t = 0; t < NT; ++t) {
            const int ox  = oxB + t * 16 + nIdx;
            const int ix  = ox * strideX + kx;
            const int ixc = (ix < Win - 1) ? ix : (Win - 1);   // clamp -> load always safe
            off[t][hv] = rowBase + ixc;
            sel[t][hv] = (ix < Win);                           // pad/OOB -> 0
        }
    }
    const int chStride = 2 * Hin * Win;     // advance 2 input channels per chunk

    v8f acc[NT];
    #pragma unroll
    for (int t = 0; t < NT; ++t) acc[t] = (v8f){};

    const v16h* wv = reinterpret_cast<const v16h*>(wsm);
    #pragma unroll 4
    for (int ch = 0; ch < nChunks; ++ch) {
        const int cbase = ch * chStride;
        // phase 1: issue all 32 gather loads (independent -> deep MLP)
        float x[NT][16];
        #pragma unroll
        for (int t = 0; t < NT; ++t)
            #pragma unroll
            for (int hv = 0; hv < 16; ++hv)
                x[t][hv] = in[cbase + off[t][hv]];
        // phase 2: A-fragment from LDS, build B-fragments, 2 WMMAs
        const v16h afrag = wv[ch * 32 + lane];
        #pragma unroll
        for (int t = 0; t < NT; ++t) {
            v16h bfrag;
            #pragma unroll
            for (int hv = 0; hv < 16; ++hv)
                bfrag[hv] = (_Float16)(sel[t][hv] ? x[t][hv] : 0.f);
            acc[t] = __builtin_amdgcn_wmma_f32_16x16x32_f16(false, afrag, false, bfrag,
                                                            (short)0, acc[t], false, false);
        }
    }

    // ---- bias + lrelu, stage activation tiles (C-layout: VGPR i -> M=i+8*laneHi) ----
    #pragma unroll
    for (int t = 0; t < NT; ++t) {
        float* atile = &act[(wid * NT + t) * 256];
        #pragma unroll
        for (int i = 0; i < 8; ++i) {
            const int m = i + laneHi * 8;
            atile[m * 16 + nIdx] = lrelu(acc[t][i] + ba[m]);
        }
    }
    __syncthreads();

    // ---- 1x1 conv via second WMMA, store HWC as 2x float4 per lane ----
    const v16h a2 = reinterpret_cast<const v16h*>(wbsm)[lane];
    #pragma unroll
    for (int t = 0; t < NT; ++t) {
        const float* atile = &act[(wid * NT + t) * 256];
        v16h b2;
        #pragma unroll
        for (int hv = 0; hv < 16; ++hv) {
            const int k = kmap(laneHi, hv);
            b2[hv] = (_Float16)((k < 16) ? atile[k * 16 + nIdx] : 0.f);
        }
        v8f c2 = {};
        c2 = __builtin_amdgcn_wmma_f32_16x16x32_f16(false, a2, false, b2,
                                                    (short)0, c2, false, false);
        const int ox = oxB + t * 16 + nIdx;
        if (tvalid && ox < Wout) {
            float* op = out + ((size_t)(oy * Wout + ox)) * 16 + laneHi * 8;
            float4 r0, r1;
            r0.x = lrelu(c2[0] + bb[laneHi * 8 + 0]);
            r0.y = lrelu(c2[1] + bb[laneHi * 8 + 1]);
            r0.z = lrelu(c2[2] + bb[laneHi * 8 + 2]);
            r0.w = lrelu(c2[3] + bb[laneHi * 8 + 3]);
            r1.x = lrelu(c2[4] + bb[laneHi * 8 + 4]);
            r1.y = lrelu(c2[5] + bb[laneHi * 8 + 5]);
            r1.z = lrelu(c2[6] + bb[laneHi * 8 + 6]);
            r1.w = lrelu(c2[7] + bb[laneHi * 8 + 7]);
            *reinterpret_cast<float4*>(op)     = r0;
            *reinterpret_cast<float4*>(op + 4) = r1;
        }
    }
}

// cv[d, y, j] = sum_c | tl[y,j,c] - (4j-d >= 0 ? tr[y,4j-d,c] : 0) |   (tl/tr are HWC)
__global__ void __launch_bounds__(256)
cost_volume_kernel(const float* __restrict__ tl, const float* __restrict__ tr,
                   int h, int w4, int w, int D, float* __restrict__ cv)
{
    const int gid = blockIdx.x * blockDim.x + threadIdx.x;
    const int total = D * h * w4;
    if (gid >= total) return;
    const int j = gid % w4;
    const int y = (gid / w4) % h;
    const int d = gid / (w4 * h);
    const int i = 4 * j - d;
    const int ic = (i > 0) ? i : 0;

    const float4* ta = reinterpret_cast<const float4*>(tl + ((size_t)(y * w4 + j)) * 16);
    const float4* tb = reinterpret_cast<const float4*>(tr + ((size_t)(y * w + ic)) * 16);
    float s = 0.f;
    #pragma unroll
    for (int q = 0; q < 4; ++q) {
        float4 a = ta[q];
        float4 b = tb[q];
        if (i < 0) { b.x = 0.f; b.y = 0.f; b.z = 0.f; b.w = 0.f; }
        s += fabsf(a.x - b.x) + fabsf(a.y - b.y) + fabsf(a.z - b.z) + fabsf(a.w - b.w);
    }
    cv[gid] = s;
}

// hyp: ch0 = argmin_d cv, ch1..2 = 0, ch3..15 = lrelu(dw * [min_cost, feat] + db)
// featHWC=1: feat is pixel-major 16-ch (tl tiles); featHWC=0: feat is CHW (raw features)
__global__ void __launch_bounds__(256)
level_hyp_kernel(const float* __restrict__ cv, const float* __restrict__ feat,
                 int Cf, int featHWC,
                 const float* __restrict__ dw, const float* __restrict__ db,
                 int h, int w4, int D, float* __restrict__ out)
{
    const int gid = blockIdx.x * blockDim.x + threadIdx.x;
    const int total = h * w4;
    if (gid >= total) return;

    float best = cv[gid];
    int bestD = 0;
    for (int d = 1; d < D; ++d) {
        const float v = cv[d * total + gid];
        if (v < best) { best = v; bestD = d; }   // first-min, matches jnp.argmin
    }
    float f[32];
    if (featHWC) {
        const float4* fp = reinterpret_cast<const float4*>(feat + (size_t)gid * 16);
        #pragma unroll
        for (int q = 0; q < 4; ++q) {
            float4 v = fp[q];
            f[q * 4 + 0] = v.x; f[q * 4 + 1] = v.y; f[q * 4 + 2] = v.z; f[q * 4 + 3] = v.w;
        }
    } else {
        for (int c = 0; c < Cf; ++c) f[c] = feat[(size_t)c * total + gid];
    }

    out[0 * total + gid] = (float)bestD;
    out[1 * total + gid] = 0.f;
    out[2 * total + gid] = 0.f;
    const int Ct = Cf + 1;
    for (int oc = 0; oc < 13; ++oc) {
        float s = db[oc] + dw[oc * Ct] * best;
        for (int c = 0; c < Cf; ++c) s += dw[oc * Ct + 1 + c] * f[c];
        out[(3 + oc) * total + gid] = lrelu(s);
    }
}

extern "C" void kernel_launch(void* const* d_in, const int* in_sizes, int n_in,
                              void* d_out, int out_size, void* d_ws, size_t ws_size,
                              hipStream_t stream)
{
    (void)in_sizes; (void)n_in; (void)out_size; (void)ws_size;
    const int H = 384, W = 1280, MAXDISP = 320;
    const int scales[5] = {16, 8, 4, 2, 1};
    const int cins[5]   = {32, 24, 24, 16, 16};

    const float* fl[5]; const float* fr[5];
    const float *wa[5], *ba[5], *wb[5], *bb[5], *dw[5], *db[5];
    for (int i = 0; i < 5; ++i) {
        fl[i] = (const float*)d_in[i];
        fr[i] = (const float*)d_in[5 + i];
        wa[i] = (const float*)d_in[10 + 4 * i];
        ba[i] = (const float*)d_in[11 + 4 * i];
        wb[i] = (const float*)d_in[12 + 4 * i];
        bb[i] = (const float*)d_in[13 + 4 * i];
        dw[i] = (const float*)d_in[30 + 2 * i];
        db[i] = (const float*)d_in[31 + 2 * i];
    }

    // workspace: tl then tr per scale (HWC layout: pixel-major, 16 channels)
    float* ws = (float*)d_ws;
    float* tl[5]; float* tr[5];
    size_t off = 0;
    int Hout[5], W4[5], Ws_[5], Dd[5];
    for (int i = 0; i < 5; ++i) {
        const int s = scales[i];
        const int Hs = H / s; Ws_[i] = W / s;
        Hout[i] = Hs / 4; W4[i] = Ws_[i] / 4; Dd[i] = MAXDISP / s;
        tl[i] = ws + off; off += (size_t)16 * Hout[i] * W4[i];
        tr[i] = ws + off; off += (size_t)16 * Hout[i] * Ws_[i];
    }

    // d_out offsets: cv16..cv1 then hyp16..hyp1 (reference layouts, CHW)
    float* outf = (float*)d_out;
    float* cvp[5]; float* hyp[5];
    size_t o = 0;
    for (int i = 0; i < 5; ++i) { cvp[i] = outf + o; o += (size_t)Dd[i] * Hout[i] * W4[i]; }
    for (int i = 0; i < 5; ++i) { hyp[i] = outf + o; o += (size_t)16 * Hout[i] * W4[i]; }

    // 1) fused conv A + conv B (WMMA) for both sides of all 5 scales
    for (int i = 0; i < 5; ++i) {
        const int s = scales[i];
        const int Hs = H / s;
        {   // left: stride (4,4), out Hout x W4
            const int tiles = Hout[i] * ((W4[i] + NT * 16 - 1) / (NT * 16));
            conv_tile_wmma<<<(tiles + NWAVES - 1) / NWAVES, 256, 0, stream>>>(
                fl[i], cins[i], Hs, Ws_[i], 4,
                wa[i], ba[i], wb[i], bb[i], tl[i], Hout[i], W4[i]);
        }
        {   // right: pad +3, stride (4,1), out Hout x Ws
            const int tiles = Hout[i] * ((Ws_[i] + NT * 16 - 1) / (NT * 16));
            conv_tile_wmma<<<(tiles + NWAVES - 1) / NWAVES, 256, 0, stream>>>(
                fr[i], cins[i], Hs, Ws_[i], 1,
                wa[i], ba[i], wb[i], bb[i], tr[i], Hout[i], Ws_[i]);
        }
    }

    // 2) cost volumes
    for (int i = 0; i < 5; ++i) {
        const int total = Dd[i] * Hout[i] * W4[i];
        cost_volume_kernel<<<(total + 255) / 256, 256, 0, stream>>>(
            tl[i], tr[i], Hout[i], W4[i], Ws_[i], Dd[i], cvp[i]);
    }

    // 3) hypotheses: feat = {tl16(HWC), tl8(HWC), fea_l16, fea_l8, fea_l4}
    const float* feats[5] = { tl[0], tl[1], fl[0], fl[1], fl[2] };
    const int    cf[5]    = { 16, 16, 32, 24, 24 };
    const int    fhwc[5]  = { 1, 1, 0, 0, 0 };
    for (int i = 0; i < 5; ++i) {
        const int total = Hout[i] * W4[i];
        level_hyp_kernel<<<(total + 255) / 256, 256, 0, stream>>>(
            cvp[i], feats[i], cf[i], fhwc[i], dw[i], db[i],
            Hout[i], W4[i], Dd[i], hyp[i]);
    }
}
